// DCNLayer_5961414607571
// MI455X (gfx1250) — matrix operand order repaired and checked by reading, not compiled, not run on hardware
//
#include <hip/hip_runtime.h>

typedef __attribute__((ext_vector_type(8)))  __bf16 v8bf;
typedef __attribute__((ext_vector_type(16))) __bf16 v16bf;
typedef __attribute__((ext_vector_type(2)))  __bf16 v2bf;
typedef __attribute__((ext_vector_type(8)))  float  v8f;
typedef __attribute__((ext_vector_type(2)))  float  v2f;

#define BATCH 4
#define CIN   64
#define HDIM  160
#define WDIM  160
#define PIX   (HDIM * WDIM)      // 25600
#define COUT  64
#define KKTAP 9
#define NCHUNK 18                // (CIN*KKTAP)/32, chunk = one tap x 32 channels
#define OFFCH 27

// ---------- float -> bf16 conversions ----------
__device__ __forceinline__ __bf16 f2bf(float f) {
    return (__bf16)f;            // clang RNE conversion
}
// pack two floats -> two bf16 in one dword; let the backend pick v_cvt_pk_bf16_f32
__device__ __forceinline__ unsigned pack2bf(float a, float b) {
#if __has_builtin(__builtin_amdgcn_cvt_pk_bf16_f32)
    v2bf r = __builtin_amdgcn_cvt_pk_bf16_f32(a, b);
#else
    v2f  f = {a, b};
    v2bf r = __builtin_convertvector(f, v2bf);
#endif
    return __builtin_bit_cast(unsigned, r);
}
__device__ __forceinline__ int iclamp(int v, int lo, int hi) {
    return v < lo ? lo : (v > hi ? hi : v);
}

// ---------- Kernel 0: NCHW -> NHWC transpose of x (per batch: C x P -> P x C) ----------
__global__ __launch_bounds__(256) void dcn_transpose_kernel(
    const float* __restrict__ x, float* __restrict__ xt)
{
    __shared__ float tile[32][33];
    const int b  = blockIdx.z;
    const int p0 = blockIdx.x * 32;   // pixel tile
    const int c0 = blockIdx.y * 32;   // channel tile
    const int tx = threadIdx.x & 31;
    const int ty = threadIdx.x >> 5;  // 0..7
    const float* xb  = x  + (size_t)b * CIN * PIX;
    float*       xtb = xt + (size_t)b * PIX * CIN;
    #pragma unroll
    for (int i = 0; i < 4; ++i) {
        const int c = c0 + ty + i * 8;
        tile[ty + i * 8][tx] = xb[(size_t)c * PIX + p0 + tx];   // coalesced read
    }
    __syncthreads();
    #pragma unroll
    for (int i = 0; i < 4; ++i) {
        const int p = p0 + ty + i * 8;
        xtb[(size_t)p * CIN + c0 + tx] = tile[tx][ty + i * 8];  // coalesced write
    }
}

// ---------- Kernel 1: repack both weight tensors into per-chunk bf16 A-fragments ----------
// wr   : [t][o(64)][kk(32)]  with t = k*2 + chHalf, kk = channel within half
// woffr: [t][m(32)][kk(32)]  rows 27..31 zero-padded
__global__ __launch_bounds__(256) void dcn_repack_kernel(
    const float* __restrict__ weight, const float* __restrict__ w_off,
    __bf16* __restrict__ wr, __bf16* __restrict__ woffr)
{
    const int idx = blockIdx.x * 256 + threadIdx.x;
    if (idx < NCHUNK * COUT * 32) {
        const int cl = idx & 31;
        const int o  = (idx >> 5) & 63;
        const int t  = idx >> 11;
        const int k  = t >> 1;
        const int c  = ((t & 1) << 5) + cl;
        wr[idx] = f2bf(weight[(o * CIN + c) * KKTAP + k]);
    }
    if (idx < NCHUNK * 32 * 32) {
        const int cl = idx & 31;
        const int m  = (idx >> 5) & 31;
        const int t  = idx >> 10;
        const int k  = t >> 1;
        const int c  = ((t & 1) << 5) + cl;
        woffr[idx] = (m < OFFCH) ? f2bf(w_off[(m * CIN + c) * KKTAP + k]) : f2bf(0.0f);
    }
}

// ---------- Kernel 2: offset conv (implicit GEMM, WMMA bf16) + sigmoid on mask ----------
__global__ __launch_bounds__(128) void dcn_offset_conv_kernel(
    const float* __restrict__ xt, const __bf16* __restrict__ woffr,
    const float* __restrict__ b_off, float* __restrict__ om)
{
    __shared__ __attribute__((aligned(32))) __bf16 lsB[2][32][40]; // double-buffered [pix][kk]
    const int tid  = threadIdx.x;
    const int b    = blockIdx.x / (PIX / 32);
    const int p0   = (blockIdx.x % (PIX / 32)) * 32;
    const int wave = tid >> 5;
    const int lane = tid & 31;
    const int wm   = wave & 1;          // M tile (rows 0..15 / 16..31)
    const int wn   = wave >> 1;         // N tile (pixels 0..15 / 16..31)
    const int cpair = tid & 15;         // gather: channel pair -> ch 2*cpair, 2*cpair+1
    const int pg    = tid >> 4;         // gather: pixel group 0..7

    // per-thread pixel geometry (tap-independent); pixels pg + j*8
    int ylist[4], xlist[4];
    #pragma unroll
    for (int j = 0; j < 4; ++j) {
        const int p = p0 + pg + (j << 3);
        ylist[j] = p / WDIM;
        xlist[j] = p - ylist[j] * WDIM;
    }
    const int cbase = b * PIX * CIN + 2 * cpair;

    v8f acc = {};
    const int mrow = lane & 15;
    const int akb  = (lane < 16) ? 0 : 16;
    const int nn   = lane & 15;
    const int bkb  = (lane < 16) ? 0 : 16;

    for (int k = 0; k < KKTAP; ++k) {
        const int ky = k / 3 - 1;
        const int kx = k % 3 - 1;
        int  offj[4];
        bool valj[4];
        #pragma unroll
        for (int j = 0; j < 4; ++j) {
            const int hy = ylist[j] + ky;
            const int wx = xlist[j] + kx;
            valj[j] = (hy >= 0) & (hy < HDIM) & (wx >= 0) & (wx < WDIM);
            offj[j] = (iclamp(hy, 0, HDIM - 1) * WDIM + iclamp(wx, 0, WDIM - 1)) * CIN;
        }
        #pragma unroll
        for (int half = 0; half < 2; ++half) {
            const int t = 2 * k + half;
            const float* bp = xt + cbase + (half << 5);
            v2f s[4];
            #pragma unroll
            for (int j = 0; j < 4; ++j)          // phase 1: issue all loads (b64, clause)
                s[j] = *(const v2f*)(bp + offj[j]);
            #pragma unroll
            for (int j = 0; j < 4; ++j) {        // phase 2: convert + LDS store (b32)
                const float vx = valj[j] ? s[j].x : 0.0f;
                const float vy = valj[j] ? s[j].y : 0.0f;
                *(unsigned*)&lsB[half][pg + (j << 3)][2 * cpair] = pack2bf(vx, vy);
            }
            __syncthreads();
            const __bf16* wp = woffr + t * (32 * 32) + (wm * 16 + mrow) * 32 + akb;
            v8bf a0 = *(const v8bf*)wp;
            v8bf a1 = *(const v8bf*)(wp + 8);
            v16bf afrag = __builtin_shufflevector(a0, a1,
                0,1,2,3,4,5,6,7,8,9,10,11,12,13,14,15);
            v16bf bfrag = *(const v16bf*)&lsB[half][wn * 16 + nn][bkb];
            acc = __builtin_amdgcn_wmma_f32_16x16x32_bf16(
                false, afrag, false, bfrag, (short)0, acc, false, false);
            // no trailing barrier: double buffering + next iteration's barrier fences reuse
        }
    }
    const int mh = (lane >> 4) << 3;    // 0 or 8
    #pragma unroll
    for (int r = 0; r < 8; ++r) {
        const int m = wm * 16 + r + mh;
        if (m < OFFCH) {
            float v = acc[r] + b_off[m];
            if (m >= 18) v = 1.0f / (1.0f + __expf(-v));   // sigmoid on mask channels
            om[(b * OFFCH + m) * PIX + p0 + wn * 16 + nn] = v;
        }
    }
}

// ---------- Kernel 3: fused bilinear gather * mask + WMMA GEMM ----------
__global__ __launch_bounds__(256) void dcn_main_kernel(
    const float* __restrict__ xt, const float* __restrict__ om,
    const __bf16* __restrict__ wr, const float* __restrict__ bias,
    float* __restrict__ out)
{
    __shared__ __attribute__((aligned(32))) __bf16 lsB[2][32][40]; // double-buffered
    const int tid  = threadIdx.x;
    const int b    = blockIdx.x / (PIX / 32);
    const int p0   = (blockIdx.x % (PIX / 32)) * 32;
    const int wave = tid >> 5;
    const int lane = tid & 31;
    const int wm   = wave & 3;          // M tile: Cout rows 16*wm..
    const int wn   = wave >> 2;         // N tile: pixels 16*wn..
    const int cpair = tid & 15;         // gather channel pair
    const int pg    = tid >> 4;         // gather pixel group 0..15

    // pixels pg, pg+16
    int ylist[2], xlist[2];
    #pragma unroll
    for (int j = 0; j < 2; ++j) {
        const int p = p0 + pg + (j << 4);
        ylist[j] = p / WDIM;
        xlist[j] = p - ylist[j] * WDIM;
    }
    const int cbase = b * PIX * CIN + 2 * cpair;

    v8f acc = {};
    const int mrow = lane & 15;
    const int akb  = (lane < 16) ? 0 : 16;
    const int nn   = lane & 15;
    const int bkb  = (lane < 16) ? 0 : 16;

    for (int k = 0; k < KKTAP; ++k) {
        const float fky = (float)(k / 3 - 1);
        const float fkx = (float)(k % 3 - 1);
        // per-tap sample geometry: mask-premultiplied bilinear weights (zeroed OOB)
        // + clamped neighbor offsets -> branchless gather
        float w00[2], w01[2], w10[2], w11[2];
        int   o00[2], o01[2], o10[2], o11[2];
        #pragma unroll
        for (int j = 0; j < 2; ++j) {
            const int p = p0 + pg + (j << 4);
            const float offy = om[(b * OFFCH + 2 * k    ) * PIX + p];
            const float offx = om[(b * OFFCH + 2 * k + 1) * PIX + p];
            const float msk  = om[(b * OFFCH + 18 + k   ) * PIX + p];
            const float py = (float)ylist[j] + fky + offy;
            const float px = (float)xlist[j] + fkx + offx;
            const float h0 = floorf(py);
            const float w0 = floorf(px);
            const float lh = py - h0;
            const float lw = px - w0;
            const int h0i = (int)h0;
            const int w0i = (int)w0;
            const bool hv0 = (h0i     >= 0) & (h0i     < HDIM);
            const bool hv1 = (h0i + 1 >= 0) & (h0i + 1 < HDIM);
            const bool wv0 = (w0i     >= 0) & (w0i     < WDIM);
            const bool wv1 = (w0i + 1 >= 0) & (w0i + 1 < WDIM);
            const int hc0 = iclamp(h0i,     0, HDIM - 1);
            const int hc1 = iclamp(h0i + 1, 0, HDIM - 1);
            const int wc0 = iclamp(w0i,     0, WDIM - 1);
            const int wc1 = iclamp(w0i + 1, 0, WDIM - 1);
            w00[j] = (hv0 & wv0) ? (1.0f - lh) * (1.0f - lw) * msk : 0.0f;
            w01[j] = (hv0 & wv1) ? (1.0f - lh) * lw          * msk : 0.0f;
            w10[j] = (hv1 & wv0) ? lh * (1.0f - lw)          * msk : 0.0f;
            w11[j] = (hv1 & wv1) ? lh * lw                   * msk : 0.0f;
            o00[j] = (hc0 * WDIM + wc0) * CIN;
            o01[j] = (hc0 * WDIM + wc1) * CIN;
            o10[j] = (hc1 * WDIM + wc0) * CIN;
            o11[j] = (hc1 * WDIM + wc1) * CIN;
        }
        #pragma unroll
        for (int half = 0; half < 2; ++half) {
            const int t = 2 * k + half;
            const float* bp = xt + cbase + (half << 5);
            v2f s00[2], s01[2], s10[2], s11[2];
            #pragma unroll
            for (int j = 0; j < 2; ++j) {        // phase 1: issue all 8 b64 loads
                s00[j] = *(const v2f*)(bp + o00[j]);
                s01[j] = *(const v2f*)(bp + o01[j]);
                s10[j] = *(const v2f*)(bp + o10[j]);
                s11[j] = *(const v2f*)(bp + o11[j]);
            }
            #pragma unroll
            for (int j = 0; j < 2; ++j) {        // phase 2: blend + pack + LDS store
                const float vx = w00[j] * s00[j].x + w01[j] * s01[j].x
                               + w10[j] * s10[j].x + w11[j] * s11[j].x;
                const float vy = w00[j] * s00[j].y + w01[j] * s01[j].y
                               + w10[j] * s10[j].y + w11[j] * s11[j].y;
                *(unsigned*)&lsB[half][pg + (j << 4)][2 * cpair] = pack2bf(vx, vy);
            }
            __syncthreads();
            const __bf16* wp = wr + t * (COUT * 32) + (wm * 16 + mrow) * 32 + akb;
            v8bf a0 = *(const v8bf*)wp;
            v8bf a1 = *(const v8bf*)(wp + 8);
            v16bf afrag = __builtin_shufflevector(a0, a1,
                0,1,2,3,4,5,6,7,8,9,10,11,12,13,14,15);
            v16bf bfrag = *(const v16bf*)&lsB[half][wn * 16 + nn][bkb];
            acc = __builtin_amdgcn_wmma_f32_16x16x32_bf16(
                false, afrag, false, bfrag, (short)0, acc, false, false);
        }
    }
    const int mh = (lane >> 4) << 3;
    #pragma unroll
    for (int r = 0; r < 8; ++r) {
        const int m = wm * 16 + r + mh;
        out[(b * COUT + m) * PIX + p0 + wn * 16 + nn] = acc[r] + bias[m];
    }
}

extern "C" void kernel_launch(void* const* d_in, const int* in_sizes, int n_in,
                              void* d_out, int out_size, void* d_ws, size_t ws_size,
                              hipStream_t stream) {
    const float* x      = (const float*)d_in[0];
    const float* w_off  = (const float*)d_in[1];
    const float* b_off  = (const float*)d_in[2];
    const float* weight = (const float*)d_in[3];
    const float* bias   = (const float*)d_in[4];
    float* out = (float*)d_out;

    // workspace layout (~37.4 MB)
    float*  xt    = (float*)d_ws;                                  // 26.2 MB NHWC x
    float*  om    = xt + (size_t)BATCH * PIX * CIN;                // 11.1 MB offsets+mask
    __bf16* wr    = (__bf16*)(om + (size_t)BATCH * OFFCH * PIX);   // 72 KB main weights
    __bf16* woffr = wr + NCHUNK * COUT * 32;                       // 36 KB offset weights

    dim3 tg(PIX / 32, CIN / 32, BATCH);
    dcn_transpose_kernel<<<tg, 256, 0, stream>>>(x, xt);
    dcn_repack_kernel<<<(NCHUNK * COUT * 32 + 255) / 256, 256, 0, stream>>>(
        weight, w_off, wr, woffr);
    dcn_offset_conv_kernel<<<BATCH * (PIX / 32), 128, 0, stream>>>(
        xt, woffr, b_off, om);
    dcn_main_kernel<<<BATCH * (PIX / 32), 256, 0, stream>>>(
        xt, om, wr, bias, out);
}
